// HeteroGN_2396591751321
// MI455X (gfx1250) — matrix operand order repaired and checked by reading, not compiled, not run on hardware
//
#include <hip/hip_runtime.h>

// ---------------------------------------------------------------------------
// HeteroGN on MI455X (gfx1250): bf16 WMMA GEMMs, f32 accumulate, f32 I/O.
//   edge:  msg = MLP_{etype}( [ef | nf[src] | nf[dst]] )   (both branches, select)
//   agg  = scatter_sum(msg, dst*2+etype)  via global_atomic_add_f32
//   node:  unf = MLP_{is_dummy}( [nf | agg] ) + nf
//   uef  = msg + ef
// Workspace: agg f32 [N*2*128] (51.2MB) + packed bf16 weights (512KB).
// ---------------------------------------------------------------------------

typedef __attribute__((ext_vector_type(16))) __bf16 v16bf;
typedef __attribute__((ext_vector_type(8)))  __bf16 v8bf;
typedef __attribute__((ext_vector_type(8)))  float  v8f;

constexpr int DM = 128;   // feature dim

// ---- A fragment (16x32 bf16, M x K) from LDS, row-major [row][k] -----------
// Lane layout: lanes 0-15: M=lane, K = ks+{0..15}
//              lanes16-31: M=lane-16, K = ks+{16..31}
static __device__ __forceinline__ v16bf load_a_frag(const __bf16* sX, int stride,
                                                    int rowBase, int ks, int lane) {
  const __bf16* p = sX + (rowBase + (lane & 15)) * stride + ks + ((lane >> 4) & 1) * 16;
  v8bf lo = *(const v8bf*)p;
  v8bf hi = *(const v8bf*)(p + 8);
  v16bf a;
#pragma unroll
  for (int i = 0; i < 8; ++i) { a[i] = lo[i]; a[i + 8] = hi[i]; }
  return a;
}

// ---- B fragment (32x16 bf16, K x N) from packed weights wt[n][k] -----------
// Lane layout: N = lane&15, K = ks + 16*(lane>=16) + i  -> 32B contiguous.
static __device__ __forceinline__ v16bf load_b_frag(const __bf16* wt, int Kdim,
                                                    int n, int ks, int lane) {
  return *(const v16bf*)(wt + (size_t)n * Kdim + ks + ((lane >> 4) & 1) * 16);
}

// ---- 16-row x 64-col GEMM strip: acc[t] += A(16xK) * W(Kx16) ---------------
template <int KD>
static __device__ __forceinline__ void gemm_strip(const __bf16* sX, int rowBase,
                                                  const __bf16* __restrict__ wt,
                                                  int t0, int lane, v8f acc[4]) {
  v8f z = {0.f, 0.f, 0.f, 0.f, 0.f, 0.f, 0.f, 0.f};
#pragma unroll
  for (int t = 0; t < 4; ++t) acc[t] = z;
#pragma unroll
  for (int ks = 0; ks < KD; ks += 32) {
    v16bf a = load_a_frag(sX, KD, rowBase, ks, lane);
    // hint next weight panel toward the caches (global_prefetch_b8)
    if (ks + 32 < KD)
      __builtin_prefetch((const void*)(wt + (size_t)(t0 * 16 + (lane & 15)) * KD + ks + 32), 0, 1);
#pragma unroll
    for (int t = 0; t < 4; ++t) {
      v16bf b = load_b_frag(wt, KD, (t0 + t) * 16 + (lane & 15), ks, lane);
      acc[t] = __builtin_amdgcn_wmma_f32_16x16x32_bf16(false, a, false, b,
                                                       (short)0, acc[t], false, false);
    }
  }
}

// ---- C tile (f32) + bias -> ReLU -> bf16 hidden in LDS [row][n] ------------
// C layout (ISA): element i -> M = i + 8*(lane>=16), N = lane&15.
static __device__ __forceinline__ void store_hidden(__bf16* sH, int rowBase, int t0,
                                                    int lane, const float* __restrict__ bias,
                                                    const v8f acc[4]) {
#pragma unroll
  for (int t = 0; t < 4; ++t) {
    int n = (t0 + t) * 16 + (lane & 15);
    float bv = bias[n];
#pragma unroll
    for (int i = 0; i < 8; ++i) {
      int r = rowBase + i + ((lane >> 4) & 1) * 8;
      sH[r * DM + n] = (__bf16)fmaxf(acc[t][i] + bv, 0.f);
    }
  }
}

// ===========================================================================
__global__ __launch_bounds__(256) void edge_kernel(
    const float* __restrict__ nf, const float* __restrict__ ef,
    const int* __restrict__ src, const int* __restrict__ dst,
    const int* __restrict__ etp,
    const __bf16* __restrict__ ow1t, const float* __restrict__ ob1,
    const __bf16* __restrict__ ow2t, const float* __restrict__ ob2,
    const __bf16* __restrict__ iw1t, const float* __restrict__ ib1,
    const __bf16* __restrict__ iw2t, const float* __restrict__ ib2,
    float* __restrict__ agg, float* __restrict__ uef, int E)
{
  __shared__ __align__(16) __bf16 smem[64 * 384 + 64 * DM];  // 48KB + 16KB = 64KB
  __bf16* sA = smem;
  __bf16* sH = smem + 64 * 384;
  const int tid  = threadIdx.x;
  const int lane = tid & 31;
  const int wave = tid >> 5;
  const int e0   = blockIdx.x * 64;

  // ---- stage A panel: [ef | nf[src] | nf[dst]] as bf16, rows past E zeroed
  for (int idx = tid; idx < 64 * 96; idx += 256) {
    int row = idx / 96, c4 = idx % 96;
    int e = e0 + row, col = c4 * 4;
    float4 v = make_float4(0.f, 0.f, 0.f, 0.f);
    if (e < E) {
      const float* p;
      if (col < 128)      p = ef + (size_t)e * DM + col;
      else if (col < 256) p = nf + (size_t)src[e] * DM + (col - 128);
      else                p = nf + (size_t)dst[e] * DM + (col - 256);
      v = *(const float4*)p;
    }
    __bf16* d = sA + row * 384 + col;
    d[0] = (__bf16)v.x; d[1] = (__bf16)v.y; d[2] = (__bf16)v.z; d[3] = (__bf16)v.w;
  }
  __syncthreads();

  const int rowBase = (wave & 3) * 16;   // 4 row-strips of 16 edges
  const int t0      = (wave >> 2) * 4;   // 2 halves of 8 n-tiles

  v8f accT[4], accO[4], accI[4];

  // outer_em branch
  gemm_strip<384>(sA, rowBase, ow1t, t0, lane, accT);
  store_hidden(sH, rowBase, t0, lane, ob1, accT);
  __syncthreads();
  gemm_strip<128>(sH, rowBase, ow2t, t0, lane, accO);
  __syncthreads();
  // inter_em branch (reuse sH)
  gemm_strip<384>(sA, rowBase, iw1t, t0, lane, accT);
  store_hidden(sH, rowBase, t0, lane, ib1, accT);
  __syncthreads();
  gemm_strip<128>(sH, rowBase, iw2t, t0, lane, accI);

  // ---- epilogue: select branch, residual, scatter-add
#pragma unroll
  for (int t = 0; t < 4; ++t) {
    int n = (t0 + t) * 16 + (lane & 15);
    float bo = ob2[n], bi = ib2[n];
#pragma unroll
    for (int i = 0; i < 8; ++i) {
      int r = rowBase + i + ((lane >> 4) & 1) * 8;
      int e = e0 + r;
      if (e < E) {
        int ty  = etp[e];
        float m = (ty == 0) ? (accO[t][i] + bo) : (accI[t][i] + bi);
        uef[(size_t)e * DM + n] = m + ef[(size_t)e * DM + n];
        unsafeAtomicAdd(&agg[((size_t)dst[e] * 2 + ty) * DM + n], m);
      }
    }
  }
}

// ===========================================================================
__global__ __launch_bounds__(256) void node_kernel(
    const float* __restrict__ nf, const float* __restrict__ agg,
    const int* __restrict__ isd,
    const __bf16* __restrict__ cw1t, const float* __restrict__ cb1,
    const __bf16* __restrict__ cw2t, const float* __restrict__ cb2,
    const __bf16* __restrict__ dw1t, const float* __restrict__ db1,
    const __bf16* __restrict__ dw2t, const float* __restrict__ db2,
    float* __restrict__ unf, int N)
{
  __shared__ __align__(16) __bf16 smem[64 * 384 + 64 * DM];
  __bf16* sA = smem;
  __bf16* sH = smem + 64 * 384;
  const int tid  = threadIdx.x;
  const int lane = tid & 31;
  const int wave = tid >> 5;
  const int v0   = blockIdx.x * 64;

  // ---- stage A panel: [nf | agg] as bf16
  for (int idx = tid; idx < 64 * 96; idx += 256) {
    int row = idx / 96, c4 = idx % 96;
    int nd = v0 + row, col = c4 * 4;
    float4 v = make_float4(0.f, 0.f, 0.f, 0.f);
    if (nd < N) {
      const float* p = (col < 128) ? (nf + (size_t)nd * DM + col)
                                   : (agg + (size_t)nd * 256 + (col - 128));
      v = *(const float4*)p;
    }
    __bf16* d = sA + row * 384 + col;
    d[0] = (__bf16)v.x; d[1] = (__bf16)v.y; d[2] = (__bf16)v.z; d[3] = (__bf16)v.w;
  }
  __syncthreads();

  const int rowBase = (wave & 3) * 16;
  const int t0      = (wave >> 2) * 4;

  v8f accT[4], accC[4], accD[4];

  gemm_strip<384>(sA, rowBase, cw1t, t0, lane, accT);
  store_hidden(sH, rowBase, t0, lane, cb1, accT);
  __syncthreads();
  gemm_strip<128>(sH, rowBase, cw2t, t0, lane, accC);
  __syncthreads();
  gemm_strip<384>(sA, rowBase, dw1t, t0, lane, accT);
  store_hidden(sH, rowBase, t0, lane, db1, accT);
  __syncthreads();
  gemm_strip<128>(sH, rowBase, dw2t, t0, lane, accD);

#pragma unroll
  for (int t = 0; t < 4; ++t) {
    int n = (t0 + t) * 16 + (lane & 15);
    float bc = cb2[n], bd = db2[n];
#pragma unroll
    for (int i = 0; i < 8; ++i) {
      int r = rowBase + i + ((lane >> 4) & 1) * 8;
      int nd = v0 + r;
      if (nd < N) {
        float m = (isd[nd] == 0) ? (accC[t][i] + bc) : (accD[t][i] + bd);
        unf[(size_t)nd * DM + n] = m + nf[(size_t)nd * DM + n];
      }
    }
  }
}

// ===========================================================================
__global__ void pack_w(const float* __restrict__ w, __bf16* __restrict__ wt,
                       int K, int Nn) {  // w[k][n] f32 -> wt[n][k] bf16
  int i = blockIdx.x * 256 + threadIdx.x;
  if (i < K * Nn) {
    int k = i / Nn, n = i % Nn;
    wt[(size_t)n * K + k] = (__bf16)w[i];
  }
}

__global__ void zero_f32(float* __restrict__ p, long n) {
  long i  = (long)blockIdx.x * blockDim.x + threadIdx.x;
  long st = (long)gridDim.x * blockDim.x;
  for (; i < n; i += st) p[i] = 0.f;
}

// ===========================================================================
extern "C" void kernel_launch(void* const* d_in, const int* in_sizes, int n_in,
                              void* d_out, int out_size, void* d_ws, size_t ws_size,
                              hipStream_t stream) {
  const float* nf  = (const float*)d_in[0];
  const float* ef  = (const float*)d_in[1];
  const int*   src = (const int*)d_in[2];
  const int*   dst = (const int*)d_in[3];
  const int*   etp = (const int*)d_in[4];
  const int*   isd = (const int*)d_in[5];
  const float* ow1 = (const float*)d_in[6];  const float* ob1 = (const float*)d_in[7];
  const float* ow2 = (const float*)d_in[8];  const float* ob2 = (const float*)d_in[9];
  const float* iw1 = (const float*)d_in[10]; const float* ib1 = (const float*)d_in[11];
  const float* iw2 = (const float*)d_in[12]; const float* ib2 = (const float*)d_in[13];
  const float* cw1 = (const float*)d_in[14]; const float* cb1 = (const float*)d_in[15];
  const float* cw2 = (const float*)d_in[16]; const float* cb2 = (const float*)d_in[17];
  const float* dw1 = (const float*)d_in[18]; const float* db1 = (const float*)d_in[19];
  const float* dw2 = (const float*)d_in[20]; const float* db2 = (const float*)d_in[21];

  const int N = in_sizes[0] / DM;
  const int E = in_sizes[1] / DM;

  // workspace layout: agg f32 [N*2*128], then packed bf16 weights
  float*  agg      = (float*)d_ws;
  size_t  aggElems = (size_t)N * 2 * DM;
  __bf16* wb   = (__bf16*)((char*)d_ws + aggElems * sizeof(float));
  __bf16* ow1t = wb;                 __bf16* iw1t = ow1t + 384 * 128;
  __bf16* cw1t = iw1t + 384 * 128;   __bf16* dw1t = cw1t + 384 * 128;
  __bf16* ow2t = dw1t + 384 * 128;   __bf16* iw2t = ow2t + 128 * 128;
  __bf16* cw2t = iw2t + 128 * 128;   __bf16* dw2t = cw2t + 128 * 128;

  const int g1 = (384 * 128 + 255) / 256, g2 = (128 * 128 + 255) / 256;
  pack_w<<<g1, 256, 0, stream>>>(ow1, ow1t, 384, 128);
  pack_w<<<g1, 256, 0, stream>>>(iw1, iw1t, 384, 128);
  pack_w<<<g1, 256, 0, stream>>>(cw1, cw1t, 384, 128);
  pack_w<<<g1, 256, 0, stream>>>(dw1, dw1t, 384, 128);
  pack_w<<<g2, 256, 0, stream>>>(ow2, ow2t, 128, 128);
  pack_w<<<g2, 256, 0, stream>>>(iw2, iw2t, 128, 128);
  pack_w<<<g2, 256, 0, stream>>>(cw2, cw2t, 128, 128);
  pack_w<<<g2, 256, 0, stream>>>(dw2, dw2t, 128, 128);

  zero_f32<<<2048, 256, 0, stream>>>(agg, (long)aggElems);

  float* unf = (float*)d_out;
  float* uef = unf + (size_t)N * DM;

  edge_kernel<<<(E + 63) / 64, 256, 0, stream>>>(nf, ef, src, dst, etp,
      ow1t, ob1, ow2t, ob2, iw1t, ib1, iw2t, ib2, agg, uef, E);
  node_kernel<<<(N + 63) / 64, 256, 0, stream>>>(nf, agg, isd,
      cw1t, cb1, cw2t, cb2, dw1t, db1, dw2t, db2, unf, N);
}
